// LSTM_74096775791316
// MI455X (gfx1250) — matrix operand order repaired and checked by reading, not compiled, not run on hardware
//
#include <hip/hip_runtime.h>
#include <hip/hip_bf16.h>

// ---------------------------------------------------------------------------
// Persistent-LSTM for MI455X (gfx1250, wave32, WMMA).
//   layer kernel: 64 persistent WGs, each owns 16 hidden units (64 gate cols).
//   Weight slice (bf16, B-operand swizzled) lives in LDS (160KB / 256KB).
//   A operand = [x_t | h_{t-1}] read from global in WMMA-A swizzled layout.
//   Per timestep: dual-accumulator 16x16x32 bf16 WMMA K-loop (no RAW NOPs) ->
//   gate exchange in LDS -> fp32 cell update (v_rcp-based activations) ->
//   swizzled h store -> device-wide barrier.
// ---------------------------------------------------------------------------

typedef unsigned short u16;
typedef __attribute__((ext_vector_type(16))) u16    v16u;
typedef __attribute__((ext_vector_type(16))) __bf16 v16bf;
typedef __attribute__((ext_vector_type(8)))  float  v8f;

#define BB   32
#define TT   512
#define NIN  256
#define HH   1024
#define NOUT 256
#define NWG  64    // persistent workgroups; each owns HH/NWG = 16 hidden units
#define KTH  32    // k-tiles covering recurrent H (1024/32)

__device__ __forceinline__ u16 f2bf(float f) {           // fp32 -> bf16 RNE
  unsigned u = __float_as_uint(f);
  u += 0x7FFFu + ((u >> 16) & 1u);
  return (u16)(u >> 16);
}
// gate activations: v_exp_f32 + v_rcp_f32 (avoid the v_div_scale fp32 divide)
__device__ __forceinline__ float sigm(float x) {
  return __builtin_amdgcn_rcpf(1.0f + __expf(-x));
}
__device__ __forceinline__ float tanh_fast(float x) {
  x = fminf(fmaxf(x, -15.0f), 15.0f);
  float e = __expf(-2.0f * x);
  return (1.0f - e) * __builtin_amdgcn_rcpf(1.0f + e);
}

__device__ __forceinline__ v8f wmma_bf16(v16u a, v16u b, v8f c) {
  return __builtin_amdgcn_wmma_f32_16x16x32_bf16(
      false, __builtin_bit_cast(v16bf, a),
      false, __builtin_bit_cast(v16bf, b),
      (short)0, c, false, false);
}

// --------------------------- setup kernels ---------------------------------

__global__ void zero_kernel(unsigned* __restrict__ p, int n) {
  for (int i = blockIdx.x * blockDim.x + threadIdx.x; i < n; i += gridDim.x * blockDim.x)
    p[i] = 0u;
}

__global__ void add_bias_kernel(const float* __restrict__ a, const float* __restrict__ b,
                                float* __restrict__ d, int n) {
  for (int i = blockIdx.x * blockDim.x + threadIdx.x; i < n; i += gridDim.x * blockDim.x)
    d[i] = a[i] + b[i];
}

// Repack W = [Wa | Wb] (row-major [N][K], fp32) into bf16 WMMA B-operand tiles:
// dst[nt][kt][lane][e], lane<16: n=nt*16+lane, k=kt*32+(e&7)+16*(e>>3) ; lane>=16: k += 8.
__global__ void repack_w_kernel(const float* __restrict__ Wa, const float* __restrict__ Wb,
                                u16* __restrict__ dst, int ntiles, int KT, int Dx, int Db) {
  long total = (long)ntiles * KT * 512;
  for (long id = (long)blockIdx.x * blockDim.x + threadIdx.x; id < total;
       id += (long)gridDim.x * blockDim.x) {
    int e = (int)(id & 15);
    int l = (int)((id >> 4) & 31);
    long rest = id >> 9;
    int kt = (int)(rest % KT);
    int nt = (int)(rest / KT);
    int n = nt * 16 + (l & 15);
    int k = kt * 32 + ((l >> 4) << 3) + (e & 7) + ((e >> 3) << 4); // B layout: lanes 16-31 hold K {8..15,24..31}
    float v = (k < Dx) ? Wa[(long)n * Dx + k] : Wb[(long)n * Db + (k - Dx)];
    dst[id] = f2bf(v);
  }
}

// x [B,T,NIN] fp32 -> bf16 WMMA A-operand tiles: dst[t][kt][mt][lane][e]
// A 16-bit layout: lane<16 row=lane holds K {0..7,16..23}; lane>=16 holds {8..15,24..31}.
__global__ void swizzle_x_kernel(const float* __restrict__ x, u16* __restrict__ dst) {
  const long total = (long)TT * 8 * 2 * 32 * 16;
  for (long id = (long)blockIdx.x * blockDim.x + threadIdx.x; id < total;
       id += (long)gridDim.x * blockDim.x) {
    int e  = (int)(id & 15);
    int l  = (int)((id >> 4) & 31);
    int mt = (int)((id >> 9) & 1);
    int kt = (int)((id >> 10) & 7);
    int t  = (int)(id >> 13);
    int b = mt * 16 + (l & 15);
    int k = kt * 32 + ((l >> 4) << 3) + (e & 7) + ((e >> 3) << 4);
    dst[id] = f2bf(x[((long)b * TT + t) * NIN + k]);
  }
}

// --------------------------- persistent LSTM layer --------------------------

template <int KTX>
__global__ void lstm_layer_kernel(const u16* __restrict__ xswz,   // [T][KTX*2*32*16] A-swizzled input
                                  const u16* __restrict__ wswz,   // [256][KT][32][16] B-swizzled [Wih|Whh]
                                  const float* __restrict__ bias, // [4H] = b_ih + b_hh
                                  const u16* __restrict__ zerobuf,// one zeroed h step
                                  u16* __restrict__ hseq,         // [T][KTH*2*32*16] A-swizzled h out
                                  unsigned* __restrict__ counter) {
  constexpr int KT    = KTX + KTH;
  constexpr int XSTEP = KTX * 2 * 32 * 16;
  constexpr int HSTEP = KTH * 2 * 32 * 16;                // 32768 bf16 per step

  extern __shared__ unsigned char smem[];
  u16*   wlds     = (u16*)smem;                                   // [4][KT][32][16]
  float* gatesbuf = (float*)(smem + (size_t)4 * KT * 512 * 2);    // [4][32][16]
  float* cbuf     = gatesbuf + 4 * 32 * 16;                       // [32][16]

  const int wg   = blockIdx.x;          // owns hidden units [wg*16, wg*16+16)
  const int lane = threadIdx.x & 31;
  const int wid  = threadIdx.x >> 5;
  const int mt   = wid & 1;             // batch tile (rows 0-15 / 16-31)
  const int gate = wid >> 1;            // 0=i 1=f 2=g 3=o

  // Stage this WG's weight slice (4 gate n-tiles x full K) into LDS once.
  for (int g = 0; g < 4; ++g) {
    const uint4* s = (const uint4*)(wswz + (size_t)(g * NWG + wg) * KT * 512);
    uint4* d = (uint4*)(wlds + (size_t)g * KT * 512);
    for (int i = threadIdx.x; i < KT * 64; i += blockDim.x) d[i] = s[i];
  }
  for (int i = threadIdx.x; i < 512; i += blockDim.x) cbuf[i] = 0.0f;
  __syncthreads();

  const float bv = bias[gate * HH + wg * 16 + (lane & 15)];
  const u16* wbase = wlds + (size_t)gate * KT * 512 + lane * 16;

  for (int t = 0; t < TT; ++t) {
    const u16* xp = xswz + (size_t)t * XSTEP + (mt * 32 + lane) * 16;
    const u16* hp = ((t == 0) ? zerobuf : (hseq + (size_t)(t - 1) * HSTEP)) +
                    (mt * 32 + lane) * 16;

    // dual accumulators: consecutive WMMAs are independent -> no RAW hazard NOPs
    v8f acc0 = {}, acc1 = {};
#pragma unroll 2
    for (int kt = 0; kt < KT; kt += 2) {
      const u16* a0 = (kt < KTX)     ? (xp + kt * 1024)       : (hp + (kt - KTX) * 1024);
      const u16* a1 = (kt + 1 < KTX) ? (xp + (kt + 1) * 1024) : (hp + (kt + 1 - KTX) * 1024);
      v16u av0 = *(const v16u*)a0;                       // global b128 x2
      v16u bv0 = *(const v16u*)(wbase + kt * 512);       // ds b128 x2
      v16u av1 = *(const v16u*)a1;
      v16u bv1 = *(const v16u*)(wbase + (kt + 1) * 512);
      acc0 = wmma_bf16(av0, bv0, acc0);
      acc1 = wmma_bf16(av1, bv1, acc1);
    }
    v8f acc = acc0 + acc1;

    // C layout: VGPR r -> M=r (lanes 0-15) / M=8+r (lanes 16-31), N=lane&15.
#pragma unroll
    for (int r = 0; r < 8; ++r) {
      int m = mt * 16 + r + ((lane >= 16) ? 8 : 0);
      gatesbuf[(gate * 32 + m) * 16 + (lane & 15)] = acc[r] + bv;
    }
    __syncthreads();

    // Cell update for this WG's 32x16 (batch x hidden) slice; h stored swizzled.
    for (int idx = threadIdx.x; idx < 512; idx += blockDim.x) {
      int b = idx >> 4, n = idx & 15;
      float ig = sigm(gatesbuf[(0 * 32 + b) * 16 + n]);
      float fg = sigm(gatesbuf[(1 * 32 + b) * 16 + n]);
      float gg = tanh_fast(gatesbuf[(2 * 32 + b) * 16 + n]);
      float og = sigm(gatesbuf[(3 * 32 + b) * 16 + n]);
      float c = fg * cbuf[idx] + ig * gg;
      cbuf[idx] = c;
      float h = og * tanh_fast(c);
      int j = wg * 16 + n;                       // global hidden index -> A swizzle
      int hkt = j >> 5, kl = j & 31;
      int hl = (b & 15) + (((kl >> 3) & 1) << 4);
      int he = (kl & 7) + (((kl >> 4) & 1) << 3);
      int hmt = b >> 4;
      hseq[(size_t)t * HSTEP + (((hkt * 2 + hmt) * 32 + hl) << 4) + he] = f2bf(h);
    }
    __threadfence();

    // Device-wide barrier: all 64 WGs must see the full new h before t+1.
    __syncthreads();
    if (threadIdx.x == 0) {
      __hip_atomic_fetch_add(counter, 1u, __ATOMIC_RELEASE, __HIP_MEMORY_SCOPE_AGENT);
      unsigned tgt = (unsigned)(t + 1) * gridDim.x;
      while (__hip_atomic_load(counter, __ATOMIC_ACQUIRE, __HIP_MEMORY_SCOPE_AGENT) < tgt)
        __builtin_amdgcn_s_sleep(1);
    }
    __syncthreads();
  }
}

// --------------------------- output projection ------------------------------

__global__ void out_proj_kernel(const u16* __restrict__ hseq1, const u16* __restrict__ wswz,
                                const float* __restrict__ bout, float* __restrict__ out) {
  const int lane = threadIdx.x & 31;
  const int wid  = threadIdx.x >> 5;
  int tile = blockIdx.x * 8 + wid;        // T*2*16 = 16384 tiles
  int t  = tile >> 5;
  int mt = (tile >> 4) & 1;
  int nt = tile & 15;

  const u16* abase = hseq1 + (size_t)t * 32768 + (mt * 32 + lane) * 16;
  const u16* bbase = wswz + ((size_t)nt * KTH * 32 + lane) * 16;

  v8f acc0 = {}, acc1 = {};
#pragma unroll 2
  for (int kt = 0; kt < KTH; kt += 2) {
    v16u av0 = *(const v16u*)(abase + kt * 1024);
    v16u bv0 = *(const v16u*)(bbase + kt * 512);
    v16u av1 = *(const v16u*)(abase + (kt + 1) * 1024);
    v16u bv1 = *(const v16u*)(bbase + (kt + 1) * 512);
    acc0 = wmma_bf16(av0, bv0, acc0);
    acc1 = wmma_bf16(av1, bv1, acc1);
  }
  v8f acc = acc0 + acc1;

  int n = nt * 16 + (lane & 15);
  float bb = bout[n];
#pragma unroll
  for (int r = 0; r < 8; ++r) {
    int b = mt * 16 + r + ((lane >= 16) ? 8 : 0);
    out[((size_t)b * TT + t) * NOUT + n] = sigm(acc[r] + bb);
  }
}

// ------------------------------- launch -------------------------------------

extern "C" void kernel_launch(void* const* d_in, const int* in_sizes, int n_in,
                              void* d_out, int out_size, void* d_ws, size_t ws_size,
                              hipStream_t stream) {
  (void)in_sizes; (void)n_in; (void)out_size; (void)ws_size; // needs ~104 MB scratch
  const float* x    = (const float*)d_in[0];
  const float* Wih0 = (const float*)d_in[1];
  const float* Whh0 = (const float*)d_in[2];
  const float* bih0 = (const float*)d_in[3];
  const float* bhh0 = (const float*)d_in[4];
  const float* Wih1 = (const float*)d_in[5];
  const float* Whh1 = (const float*)d_in[6];
  const float* bih1 = (const float*)d_in[7];
  const float* bhh1 = (const float*)d_in[8];
  const float* Wout = (const float*)d_in[9];
  const float* bout = (const float*)d_in[10];
  float* out = (float*)d_out;

  char* ws = (char*)d_ws;
  size_t off = 0;
  auto take = [&](size_t bytes) -> char* {
    char* p = ws + off;
    off = (off + bytes + 255) & ~(size_t)255;
    return p;
  };
  unsigned* counters = (unsigned*)take(256);                    // 2 barrier counters
  u16* zerobuf = (u16*)take((size_t)32768 * 2);                 // contiguous after counters
  u16* wswz0   = (u16*)take((size_t)256 * 40 * 512 * 2);        // [Wih0|Whh0], K=1280
  u16* wswz1   = (u16*)take((size_t)256 * 64 * 512 * 2);        // [Wih1|Whh1], K=2048
  u16* woutswz = (u16*)take((size_t)16 * 32 * 512 * 2);         // Wout, K=1024
  float* bias0 = (float*)take(4096 * 4);
  float* bias1 = (float*)take(4096 * 4);
  u16* xswz    = (u16*)take((size_t)TT * 8192 * 2);             // x, A-swizzled
  u16* hseq0   = (u16*)take((size_t)TT * 32768 * 2);            // layer-0 h, A-swizzled
  u16* hseq1   = (u16*)take((size_t)TT * 32768 * 2);            // layer-1 h, A-swizzled

  // counters + zerobuf are contiguous: zero both in one pass.
  zero_kernel<<<64, 256, 0, stream>>>(counters, (256 + 65536) / 4);

  repack_w_kernel<<<512, 256, 0, stream>>>(Wih0, Whh0, wswz0, 256, 40, NIN, HH);
  repack_w_kernel<<<512, 256, 0, stream>>>(Wih1, Whh1, wswz1, 256, 64, HH, HH);
  repack_w_kernel<<<64, 256, 0, stream>>>(Wout, Wout, woutswz, 16, 32, HH, HH);
  add_bias_kernel<<<16, 256, 0, stream>>>(bih0, bhh0, bias0, 4096);
  add_bias_kernel<<<16, 256, 0, stream>>>(bih1, bhh1, bias1, 4096);
  swizzle_x_kernel<<<512, 256, 0, stream>>>(x, xswz);

  const size_t shm0 = (size_t)4 * 40 * 512 * 2 + 4 * 32 * 16 * 4 + 32 * 16 * 4; // 174,080 B
  const size_t shm1 = (size_t)4 * 64 * 512 * 2 + 4 * 32 * 16 * 4 + 32 * 16 * 4; // 272,384 B
  (void)hipFuncSetAttribute((const void*)lstm_layer_kernel<8>,
                            hipFuncAttributeMaxDynamicSharedMemorySize, (int)shm0);
  (void)hipFuncSetAttribute((const void*)lstm_layer_kernel<32>,
                            hipFuncAttributeMaxDynamicSharedMemorySize, (int)shm1);

  lstm_layer_kernel<8><<<NWG, 256, shm0, stream>>>(xswz, wswz0, bias0, zerobuf, hseq0,
                                                   counters + 0);
  lstm_layer_kernel<32><<<NWG, 256, shm1, stream>>>(hseq0, wswz1, bias1, zerobuf, hseq1,
                                                    counters + 1);
  out_proj_kernel<<<2048, 256, 0, stream>>>(hseq1, woutswz, bout, out);
}
